// MultiHeadDifferAttention_79328045957412
// MI455X (gfx1250) — matrix operand order repaired and checked by reading, not compiled, not run on hardware
//
#include <hip/hip_runtime.h>

typedef __attribute__((ext_vector_type(16))) __bf16 v16bf;
typedef __attribute__((ext_vector_type(8)))  float  v8f;
typedef __attribute__((ext_vector_type(4)))  float  f32x4;
typedef __attribute__((ext_vector_type(4)))  int    i32x4;

#define DIMC 2048
#define SEQ  2048
#define NH   16

// CDNA5 async global->LDS staging (ASYNCcnt path), with safe fallback.
#if defined(__gfx1250__) && __has_builtin(__builtin_amdgcn_global_load_async_to_lds_b128) && \
    __has_builtin(__builtin_amdgcn_s_wait_asynccnt)
#define USE_ASYNC_LDS 1
#define ASYNC_CP16(g, l)                                              \
  __builtin_amdgcn_global_load_async_to_lds_b128(                     \
      (__attribute__((address_space(1))) i32x4*)(g),                  \
      (__attribute__((address_space(3))) i32x4*)(l), 0, 0)
#define ASYNC_WAIT() __builtin_amdgcn_s_wait_asynccnt(0)
#else
#define USE_ASYNC_LDS 0
#endif

__device__ __forceinline__ unsigned short f2bf_raw(float f) {
  unsigned u = __builtin_bit_cast(unsigned, f);
  unsigned r = u + 0x7FFFu + ((u >> 16) & 1u);
  return (unsigned short)(r >> 16);
}
__device__ __forceinline__ __bf16 f2bf(float f) {
  return __builtin_bit_cast(__bf16, f2bf_raw(f));
}
__device__ __forceinline__ float bf2f(__bf16 h) {
  unsigned u = ((unsigned)__builtin_bit_cast(unsigned short, h)) << 16;
  return __builtin_bit_cast(float, u);
}

union FragBf { v16bf v; f32x4 f[2]; };

// ---------------------------------------------------------------------------
// W (f32, [K][N]) -> Wt (bf16, [N][K])
// ---------------------------------------------------------------------------
__global__ void wtrans_kernel(const float* __restrict__ W, __bf16* __restrict__ Wt) {
  __shared__ float tile[32][33];
  int t = threadIdx.x;
  int tx = t & 31, ty = t >> 5;
  int n0 = blockIdx.x * 32, k0 = blockIdx.y * 32;
#pragma unroll
  for (int j = 0; j < 4; ++j)
    tile[ty + 8 * j][tx] = W[(size_t)(k0 + ty + 8 * j) * DIMC + n0 + tx];
  __syncthreads();
#pragma unroll
  for (int j = 0; j < 4; ++j)
    Wt[(size_t)(n0 + ty + 8 * j) * DIMC + k0 + tx] = f2bf(tile[tx][ty + 8 * j]);
}

// ---------------------------------------------------------------------------
// C[4096 x 2048] = X[4096 x 2048](f32->bf16) @ W (via Wt[N][K] bf16) + bias
// block = 256 thr (8 waves), tile 128x128, k-chunk 32, WMMA bf16 16x16x32
// ---------------------------------------------------------------------------
__global__ void __launch_bounds__(256)
gemm_kernel(const float* __restrict__ X, const __bf16* __restrict__ Wt,
            const float* __restrict__ bias, __bf16* __restrict__ Cout) {
  __shared__ __bf16 As[128][40];  // [m][k]
  __shared__ __bf16 Bs[128][40];  // [n][k]
  int t = threadIdx.x;
  int wv = t >> 5, lane = t & 31, lh = lane >> 4, ln = lane & 15;
  int m0 = blockIdx.y * 128, n0 = blockIdx.x * 128;
  int mb = (wv >> 1) * 32, nb = (wv & 1) * 64;

  const v8f vzero = {0.f, 0.f, 0.f, 0.f, 0.f, 0.f, 0.f, 0.f};
  v8f acc[2][4];
#pragma unroll
  for (int mt = 0; mt < 2; ++mt)
#pragma unroll
    for (int nt = 0; nt < 4; ++nt) acc[mt][nt] = vzero;

  int lrow = t & 127, lhalf = t >> 7;
  for (int kc = 0; kc < DIMC; kc += 32) {
    // stage Wt (already bf16): pure copy -> async DMA to LDS when available
    const __bf16* wsrc = Wt + (size_t)(n0 + lrow) * DIMC + kc + lhalf * 16;
#if USE_ASYNC_LDS
    ASYNC_CP16(wsrc, &Bs[lrow][lhalf * 16]);
    ASYNC_CP16(wsrc + 8, &Bs[lrow][lhalf * 16 + 8]);
#else
    {
      f32x4* bdst = (f32x4*)&Bs[lrow][lhalf * 16];
      bdst[0] = ((const f32x4*)wsrc)[0];
      bdst[1] = ((const f32x4*)wsrc)[1];
    }
#endif
    // stage X (convert f32 -> bf16, must pass through VGPRs)
    const f32x4* xsrc = (const f32x4*)(X + (size_t)(m0 + lrow) * DIMC + kc + lhalf * 16);
    unsigned* adst = (unsigned*)&As[lrow][lhalf * 16];
#pragma unroll
    for (int j = 0; j < 4; ++j) {
      f32x4 xv = xsrc[j];
      adst[2 * j]     = (unsigned)f2bf_raw(xv.x) | ((unsigned)f2bf_raw(xv.y) << 16);
      adst[2 * j + 1] = (unsigned)f2bf_raw(xv.z) | ((unsigned)f2bf_raw(xv.w) << 16);
    }
#if USE_ASYNC_LDS
    ASYNC_WAIT();
#endif
    __syncthreads();

    FragBf a[2], bf[4];
#pragma unroll
    for (int mt = 0; mt < 2; ++mt) {
      int r = mb + mt * 16 + ln;
      a[mt].f[0] = *(const f32x4*)&As[r][8 * lh];
      a[mt].f[1] = *(const f32x4*)&As[r][16 + 8 * lh];
    }
#pragma unroll
    for (int nt = 0; nt < 4; ++nt) {
      int r = nb + nt * 16 + ln;
      bf[nt].f[0] = *(const f32x4*)&Bs[r][8 * lh];
      bf[nt].f[1] = *(const f32x4*)&Bs[r][16 + 8 * lh];
    }
#pragma unroll
    for (int mt = 0; mt < 2; ++mt)
#pragma unroll
      for (int nt = 0; nt < 4; ++nt)
        acc[mt][nt] = __builtin_amdgcn_wmma_f32_16x16x32_bf16(
            false, a[mt].v, false, bf[nt].v, (short)0, acc[mt][nt], false, false);
    __syncthreads();
  }

#pragma unroll
  for (int mt = 0; mt < 2; ++mt)
#pragma unroll
    for (int nt = 0; nt < 4; ++nt) {
      int cg = n0 + nb + nt * 16 + ln;
      float bv = bias[cg];
#pragma unroll
      for (int i = 0; i < 8; ++i) {
        int rg = m0 + mb + mt * 16 + i + 8 * lh;
        Cout[(size_t)rg * DIMC + cg] = f2bf(acc[mt][nt][i] + bv);
      }
    }
}

// ---------------------------------------------------------------------------
// Rotary: raw [b,s, h*64+d] bf16 -> dst [b, h32, s, 64] bf16 (q scaled)
// ---------------------------------------------------------------------------
__global__ void rotary_kernel(const __bf16* __restrict__ raw,
                              const float* __restrict__ cosb,
                              const float* __restrict__ sinb,
                              __bf16* __restrict__ dst, float scale) {
  int tid = blockIdx.x * blockDim.x + threadIdx.x;  // 2*2048*32*32
  int j = tid & 31;
  int h = (tid >> 5) & 31;
  int s = (tid >> 10) & 2047;
  int b = tid >> 21;
  size_t src = (size_t)(b * SEQ + s) * DIMC + h * 64 + 2 * j;
  float x1 = bf2f(raw[src]), x2 = bf2f(raw[src + 1]);
  float c = cosb[s * 32 + j], sn = sinb[s * 32 + j];
  float o1 = (x1 * c - x2 * sn) * scale;
  float o2 = (x1 * sn + x2 * c) * scale;
  size_t d = ((size_t)((b * 32 + h) * SEQ + s)) * 64 + 2 * j;
  dst[d] = f2bf(o1);
  dst[d + 1] = f2bf(o2);
}

// ---------------------------------------------------------------------------
// V transpose: vraw [b,s, h*128+d] bf16 -> Vt [b, h, d(128), s] bf16
// ---------------------------------------------------------------------------
__global__ void vtrans_kernel(const __bf16* __restrict__ vraw, __bf16* __restrict__ Vt) {
  int tid = blockIdx.x * blockDim.x + threadIdx.x;  // 2*16*128*2048
  int s = tid & 2047;
  int d = (tid >> 11) & 127;
  int h = (tid >> 18) & 15;
  int b = tid >> 22;
  Vt[((size_t)((b * NH + h) * 128 + d)) * SEQ + s] =
      vraw[(size_t)(b * SEQ + s) * DIMC + h * 128 + d];
}

// ---------------------------------------------------------------------------
// lambda = exp(lq1.lk1) - exp(lq2.lk2) + LAMBDA_INIT
// ---------------------------------------------------------------------------
__global__ void lambda_kernel(const float* __restrict__ lq1, const float* __restrict__ lk1,
                              const float* __restrict__ lq2, const float* __restrict__ lk2,
                              float* __restrict__ lam) {
  __shared__ float s1[64], s2[64];
  int t = threadIdx.x;
  s1[t] = lq1[t] * lk1[t];
  s2[t] = lq2[t] * lk2[t];
  __syncthreads();
  if (t == 0) {
    float d1 = 0.f, d2 = 0.f;
    for (int i = 0; i < 64; ++i) { d1 += s1[i]; d2 += s2[i]; }
    *lam = __expf(d1) - __expf(d2) + 0.7836057665316245f;
  }
}

// ---------------------------------------------------------------------------
// Flash differential attention.
// grid: (qtile=32, h=16, b=2); block 256 = 8 waves = 4 q-subtiles x 2 parities.
// Q pre-scaled. out = softmax(Q0 K0^T + mask) V - lam * softmax(Q1 K1^T + mask) V
// ---------------------------------------------------------------------------
__global__ void __launch_bounds__(256)
flash_kernel(const __bf16* __restrict__ Qb, const __bf16* __restrict__ Kb,
             const __bf16* __restrict__ Vtb, const float* __restrict__ lamp,
             float* __restrict__ out) {
  __shared__ __bf16 Ks[2][32][64];    // [parity][key][d]      8 KB
  __shared__ __bf16 Vs[128][32];      // [vdim][key]           8 KB
  __shared__ __bf16 Ps[8][16][32];    // per-wave P staging    8 KB
  __shared__ float  Os[4][16][128];   // parity-1 combine     32 KB

  int t = threadIdx.x;
  int wv = t >> 5, lane = t & 31, lh = lane >> 4, ln = lane & 15;
  int qt = wv >> 1, par = wv & 1;
  int qtile = blockIdx.x, h = blockIdx.y, b = blockIdx.z;
  int q0w = qtile * 64 + qt * 16;
  int h32 = 2 * h + par;

  // Q fragments (A layout), loaded directly from global
  const __bf16* Qg = Qb + ((size_t)(b * 32 + h32) * SEQ + q0w + ln) * 64;
  FragBf aq[2];
#pragma unroll
  for (int dcb = 0; dcb < 2; ++dcb) {
    aq[dcb].f[0] = *(const f32x4*)(Qg + dcb * 32 + 8 * lh);
    aq[dcb].f[1] = *(const f32x4*)(Qg + dcb * 32 + 16 + 8 * lh);
  }

  const v8f vzero = {0.f, 0.f, 0.f, 0.f, 0.f, 0.f, 0.f, 0.f};
  v8f o[8];
#pragma unroll
  for (int nt = 0; nt < 8; ++nt) o[nt] = vzero;
  float m[8], l[8];
#pragma unroll
  for (int i = 0; i < 8; ++i) { m[i] = -1e30f; l[i] = 0.f; }

  const __bf16* Kg = Kb + (size_t)(b * 32 + 2 * h) * SEQ * 64;
  const __bf16* Vg = Vtb + (size_t)(b * NH + h) * 128 * SEQ;

  int kvend = qtile * 64 + 63;
  for (int kv0 = 0; kv0 <= kvend; kv0 += 32) {
    // cooperative staging (both parities' K, shared V^T)
#if USE_ASYNC_LDS
#pragma unroll
    for (int it = 0; it < 2; ++it) {
      int idx = t + it * 256;
      int p = idx >> 8, row = (idx >> 3) & 31, seg = idx & 7;
      ASYNC_CP16(Kg + ((size_t)p * SEQ + kv0 + row) * 64 + seg * 8, &Ks[p][row][seg * 8]);
    }
#pragma unroll
    for (int it = 0; it < 2; ++it) {
      int idx = t + it * 256;
      int row = idx >> 2, seg = idx & 3;
      ASYNC_CP16(Vg + (size_t)row * SEQ + kv0 + seg * 8, &Vs[row][seg * 8]);
    }
    ASYNC_WAIT();
#else
#pragma unroll
    for (int it = 0; it < 2; ++it) {
      int idx = t + it * 256;
      int p = idx >> 8, row = (idx >> 3) & 31, seg = idx & 7;
      *(f32x4*)&Ks[p][row][seg * 8] =
          *(const f32x4*)(Kg + ((size_t)p * SEQ + kv0 + row) * 64 + seg * 8);
    }
#pragma unroll
    for (int it = 0; it < 2; ++it) {
      int idx = t + it * 256;
      int row = idx >> 2, seg = idx & 3;
      *(f32x4*)&Vs[row][seg * 8] =
          *(const f32x4*)(Vg + (size_t)row * SEQ + kv0 + seg * 8);
    }
#endif
    __syncthreads();

    if (kv0 <= q0w + 15) {  // wave-uniform; EXEC stays full
      // S = Q K^T : two 16x16 tiles over 32 keys, K-dim 64 = 2 chunks
      v8f s0 = vzero, s1 = vzero;
#pragma unroll
      for (int dcb = 0; dcb < 2; ++dcb) {
        FragBf b0, b1;
        b0.f[0] = *(const f32x4*)&Ks[par][ln][dcb * 32 + 8 * lh];
        b0.f[1] = *(const f32x4*)&Ks[par][ln][dcb * 32 + 16 + 8 * lh];
        b1.f[0] = *(const f32x4*)&Ks[par][16 + ln][dcb * 32 + 8 * lh];
        b1.f[1] = *(const f32x4*)&Ks[par][16 + ln][dcb * 32 + 16 + 8 * lh];
        s0 = __builtin_amdgcn_wmma_f32_16x16x32_bf16(false, aq[dcb].v, false, b0.v,
                                                     (short)0, s0, false, false);
        s1 = __builtin_amdgcn_wmma_f32_16x16x32_bf16(false, aq[dcb].v, false, b1.v,
                                                     (short)0, s1, false, false);
      }

      // causal mask + online softmax (row M = i + 8*lh, N = ln / 16+ln)
      float p0a[8], p1a[8], scl[8];
#pragma unroll
      for (int i = 0; i < 8; ++i) {
        int qrow = q0w + i + 8 * lh;
        float v0 = s0[i], v1 = s1[i];
        if (kv0 + ln > qrow) v0 = -1e30f;
        if (kv0 + 16 + ln > qrow) v1 = -1e30f;
        float rm = fmaxf(v0, v1);
#pragma unroll
        for (int off = 1; off < 16; off <<= 1) rm = fmaxf(rm, __shfl_xor(rm, off, 32));
        float mn = fmaxf(m[i], rm);
        float sc = __expf(m[i] - mn);
        float e0 = __expf(v0 - mn);
        float e1 = __expf(v1 - mn);
        float rs = e0 + e1;
#pragma unroll
        for (int off = 1; off < 16; off <<= 1) rs += __shfl_xor(rs, off, 32);
        l[i] = l[i] * sc + rs;
        m[i] = mn;
        scl[i] = sc;
        p0a[i] = e0;
        p1a[i] = e1;
      }
#pragma unroll
      for (int nt = 0; nt < 8; ++nt)
#pragma unroll
        for (int i = 0; i < 8; ++i) o[nt][i] *= scl[i];

      // restage P (C layout -> A layout) through per-wave LDS
#pragma unroll
      for (int i = 0; i < 8; ++i) {
        Ps[wv][i + 8 * lh][ln] = f2bf(p0a[i]);
        Ps[wv][i + 8 * lh][16 + ln] = f2bf(p1a[i]);
      }
      FragBf pf;
      pf.f[0] = *(const f32x4*)&Ps[wv][ln][8 * lh];
      pf.f[1] = *(const f32x4*)&Ps[wv][ln][16 + 8 * lh];

      // O += P @ V  (8 n-tiles of 16 v-dims)
#pragma unroll
      for (int nt = 0; nt < 8; ++nt) {
        FragBf bv;
        bv.f[0] = *(const f32x4*)&Vs[nt * 16 + ln][8 * lh];
        bv.f[1] = *(const f32x4*)&Vs[nt * 16 + ln][16 + 8 * lh];
        o[nt] = __builtin_amdgcn_wmma_f32_16x16x32_bf16(false, pf.v, false, bv.v,
                                                        (short)0, o[nt], false, false);
      }
    }
    __syncthreads();
  }

  // normalize
  float inv[8];
#pragma unroll
  for (int i = 0; i < 8; ++i) inv[i] = 1.f / l[i];
#pragma unroll
  for (int nt = 0; nt < 8; ++nt)
#pragma unroll
    for (int i = 0; i < 8; ++i) o[nt][i] *= inv[i];

  float lam = lamp[0];
  if (par == 1) {
#pragma unroll
    for (int nt = 0; nt < 8; ++nt)
#pragma unroll
      for (int i = 0; i < 8; ++i) Os[qt][i + 8 * lh][nt * 16 + ln] = o[nt][i];
  }
  __syncthreads();
  if (par == 0) {
#pragma unroll
    for (int nt = 0; nt < 8; ++nt)
#pragma unroll
      for (int i = 0; i < 8; ++i) {
        int srow = q0w + i + 8 * lh;
        int col = nt * 16 + ln;
        out[((size_t)(b * SEQ + srow) * NH + h) * 128 + col] =
            o[nt][i] - lam * Os[qt][i + 8 * lh][col];
      }
  }
}

// ---------------------------------------------------------------------------
extern "C" void kernel_launch(void* const* d_in, const int* in_sizes, int n_in,
                              void* d_out, int out_size, void* d_ws, size_t ws_size,
                              hipStream_t stream) {
  const float* x    = (const float*)d_in[0];
  const float* fcos = (const float*)d_in[1];
  const float* fsin = (const float*)d_in[2];
  const float* Wq   = (const float*)d_in[3];
  const float* bq   = (const float*)d_in[4];
  const float* Wk   = (const float*)d_in[5];
  const float* bk   = (const float*)d_in[6];
  const float* Wv   = (const float*)d_in[7];
  const float* bv   = (const float*)d_in[8];
  const float* lq1  = (const float*)d_in[9];
  const float* lk1  = (const float*)d_in[10];
  const float* lq2  = (const float*)d_in[11];
  const float* lk2  = (const float*)d_in[12];
  float* out = (float*)d_out;

  char* ws = (char*)d_ws;
  size_t off = 0;
  auto take = [&](size_t bytes) {
    char* p = ws + off;
    off = (off + bytes + 255) & ~(size_t)255;
    return p;
  };
  const size_t WT_B  = (size_t)DIMC * DIMC * 2;       // 8 MB
  const size_t RAW_B = (size_t)4096 * DIMC * 2;       // 16 MB
  const size_t QK_B  = (size_t)2 * 32 * SEQ * 64 * 2; // 16 MB
  const size_t VT_B  = (size_t)2 * NH * 128 * SEQ * 2;

  __bf16* Wtq  = (__bf16*)take(WT_B);
  __bf16* Wtk  = (__bf16*)take(WT_B);
  __bf16* Wtv  = (__bf16*)take(WT_B);
  __bf16* qraw = (__bf16*)take(RAW_B);
  __bf16* kraw = (__bf16*)take(RAW_B);
  __bf16* vraw = (__bf16*)take(RAW_B);
  __bf16* Qbuf = (__bf16*)take(QK_B);
  __bf16* Kbuf = (__bf16*)take(QK_B);
  __bf16* Vtb  = (__bf16*)take(VT_B);
  float*  lamp = (float*)take(256);

  dim3 tgrid(64, 64);
  wtrans_kernel<<<tgrid, 256, 0, stream>>>(Wq, Wtq);
  wtrans_kernel<<<tgrid, 256, 0, stream>>>(Wk, Wtk);
  wtrans_kernel<<<tgrid, 256, 0, stream>>>(Wv, Wtv);
  lambda_kernel<<<1, 64, 0, stream>>>(lq1, lk1, lq2, lk2, lamp);

  dim3 ggrid(DIMC / 128, 4096 / 128);
  gemm_kernel<<<ggrid, 256, 0, stream>>>(x, Wtq, bq, qraw);
  gemm_kernel<<<ggrid, 256, 0, stream>>>(x, Wtk, bk, kraw);
  gemm_kernel<<<ggrid, 256, 0, stream>>>(x, Wtv, bv, vraw);

  rotary_kernel<<<(2 * SEQ * 32 * 32) / 256, 256, 0, stream>>>(qraw, fcos, fsin, Qbuf, 0.125f);
  rotary_kernel<<<(2 * SEQ * 32 * 32) / 256, 256, 0, stream>>>(kraw, fcos, fsin, Kbuf, 1.0f);
  vtrans_kernel<<<(2 * NH * 128 * SEQ) / 256, 256, 0, stream>>>(vraw, Vtb);

  dim3 fgrid(SEQ / 64, NH, 2);
  flash_kernel<<<fgrid, 256, 0, stream>>>(Qbuf, Kbuf, Vtb, lamp, out);

  (void)in_sizes; (void)n_in; (void)out_size; (void)ws_size;
}
